// Actor_77730318123180
// MI455X (gfx1250) — matrix operand order repaired and checked by reading, not compiled, not run on hardware
//
#include <hip/hip_runtime.h>
#include <hip/hip_bf16.h>

typedef _Float16 v16h __attribute__((ext_vector_type(16)));
typedef _Float16 v8h  __attribute__((ext_vector_type(8)));
typedef float    v8f  __attribute__((ext_vector_type(8)));

constexpr int kB = 64;       // batch
constexpr int kT = 1000;     // timesteps
constexpr int kI = 128;      // input dim
constexpr int kH = 512;      // hidden dim
constexpr int kA = 16;       // action dim
constexpr int kHp = kH + 8;  // padded LDS row stride in halves (breaks 64-bank
                             // conflicts: 520*2/4 = 260 dwords, 260 % 64 = 4)

union AFrag { v16h v; v8h h[2]; };

static __device__ __forceinline__ v8f wmma_f16(v16h a, v16h b, v8f c) {
  // D = A(16x32 f16) * B(32x16 f16) + C(16x16 f32)
  return __builtin_amdgcn_wmma_f32_16x16x32_f16(
      /*neg_a=*/false, a, /*neg_b=*/false, b,
      /*c_mod=*/(short)0, c, /*reuse_a=*/false, /*reuse_b=*/false);
}

// ---------------------------------------------------------------------------
// Kernel 0: convert weights to f16 in *WMMA B-fragment tile order*.
// B-fragment rule (ISA 7.12.2): lane l holds column n=l%16 of the 32x16 tile as 16
// k values. Storing W as wtile[n_tile][k][16] makes each fragment
// load a 1KB-contiguous-per-wave coalesced burst.
//   wih16t : [kH/16][kI][16] f16   (tiled W_ih)
//   whh16t : [kH/16][kH][16] f16   (tiled W_hh)
//   mwT16  : [kH][kA] f16          (mean_w^T; already tile order, N=16)
//   swT16  : [kH][kA] f16          (std_w^T)
// ---------------------------------------------------------------------------
__global__ __launch_bounds__(256) void convert_kernel(
    const float* __restrict__ wih, const float* __restrict__ whh,
    const float* __restrict__ mw,  const float* __restrict__ sw,
    _Float16* __restrict__ wih16t, _Float16* __restrict__ whh16t,
    _Float16* __restrict__ mwT16,  _Float16* __restrict__ swT16) {
  int i = blockIdx.x * 256 + threadIdx.x;
  if (i < kI * kH) {
    int k = i >> 9, n = i & (kH - 1);
    wih16t[((n >> 4) * kI + k) * 16 + (n & 15)] = (_Float16)wih[i];
  }
  if (i < kH * kH) {
    int k = i >> 9, n = i & (kH - 1);
    whh16t[((n >> 4) * kH + k) * 16 + (n & 15)] = (_Float16)whh[i];
  }
  if (i < kH * kA) {
    int a = i & (kA - 1);
    int k = i >> 4;
    mwT16[k * kA + a] = (_Float16)mw[a * kH + k];
    swT16[k * kA + a] = (_Float16)sw[a * kH + k];
  }
}

// ---------------------------------------------------------------------------
// Kernel 1: XP = x @ W_ih, written into the all_hs output region.
// M = B*T = 64000 rows, K = 128, N = 512. One 16x16 tile per wave.
// ---------------------------------------------------------------------------
__global__ __launch_bounds__(256) void xproj_kernel(
    const float* __restrict__ x, const _Float16* __restrict__ wih16t,
    float* __restrict__ xp /* == all_hs region, [B*T][kH] */) {
  int gw   = (blockIdx.x * 256 + threadIdx.x) >> 5;  // global wave id
  int lane = threadIdx.x & 31;
  int mtile = gw >> 5;        // 32 n-tiles per m-tile
  int ntile = gw & 31;
  if (mtile >= (kB * kT) / 16) return;
  int m0 = mtile * 16, n0 = ntile * 16;
  int r  = m0 + (lane & 15);
  int hi = lane >> 4;

  v8f c = {};
  #pragma unroll
  for (int kk = 0; kk < kI; kk += 32) {
    AFrag af;
    const float* arow = x + r * kI + kk + hi * 8;
    #pragma unroll
    for (int e = 0; e < 8; ++e) af.h[0][e] = (_Float16)arow[e];
    #pragma unroll
    for (int e = 0; e < 8; ++e) af.h[1][e] = (_Float16)arow[16 + e];
    // tiled B fragment: lane holds column n = lane&15, k in A's table order
    v16h bf;
    #pragma unroll
    for (int s = 0; s < 16; ++s)
      bf[s] = wih16t[(ntile * kI + kk + (s & 7) + hi * 8 + (s >> 3) * 16) * 16 + (lane & 15)];
    c = wmma_f16(af.v, bf, c);
  }
  int Mrow = hi * 8, Ncol = lane & 15;
  #pragma unroll
  for (int v = 0; v < 8; ++v)
    xp[(m0 + Mrow + v) * kH + n0 + Ncol] = c[v];
}

// ---------------------------------------------------------------------------
// Kernel 2: the serial recurrence. ONE workgroup, 32 waves (1024 threads).
//   g = f16(y*h) in LDS with padded rows (bank-conflict-free A fragments).
//   y state in registers (exclusive tile ownership, 32 VGPRs/lane).
//   W_hh f16 (tiled) streams coalesced from L2 each step (512 KB resident).
//   all_hs arrives pre-filled with XP; read XP[b,t] as WMMA C, overwrite
//   the same slot with h_t. Two s_barriers per step.
// ---------------------------------------------------------------------------
__global__ __launch_bounds__(1024) void rnn_recurrence_kernel(
    const float* __restrict__ hn, const float* __restrict__ yd0,
    const _Float16* __restrict__ whh16t,
    float* __restrict__ all_hs,  // [B][T][H], pre-filled with XP
    float* __restrict__ h_last, float* __restrict__ y_last) {
  __shared__ _Float16 g[kB * kHp];  // 66,560 bytes (padded rows)

  int tid  = threadIdx.x;
  int wave = tid >> 5;
  int lane = tid & 31;
  int rt = wave & 3;   // row tile: batch rows rt*16 .. rt*16+15
  int cg = wave >> 2;  // column group: tiles cg*4 .. cg*4+3 (cols cg*64..+63)
  int Mrow = (lane >> 4) * 8;
  int Ncol = lane & 15;

  // init g = f16(y0 * h0), cooperative
  for (int i = tid; i < kB * kH; i += 1024) {
    int row = i >> 9, col = i & (kH - 1);
    g[row * kHp + col] = (_Float16)(hn[i] * yd0[i]);
  }

  // per-lane y state for the 4 owned tiles
  float yreg[4][8];
  #pragma unroll
  for (int j = 0; j < 4; ++j) {
    int n0 = (cg * 4 + j) * 16;
    #pragma unroll
    for (int v = 0; v < 8; ++v) {
      int b = rt * 16 + Mrow + v;
      yreg[j][v] = yd0[b * kH + n0 + Ncol];
    }
  }
  __syncthreads();

  int arow  = rt * 16 + (lane & 15);
  int ahalf = (lane >> 4) * 8;

  for (int t = 0; t < kT; ++t) {
    // C init = XP tile (read from all_hs slot we will overwrite)
    v8f c[4];
    #pragma unroll
    for (int j = 0; j < 4; ++j) {
      int n0 = (cg * 4 + j) * 16;
      #pragma unroll
      for (int v = 0; v < 8; ++v) {
        int b = rt * 16 + Mrow + v;
        c[j][v] = all_hs[(b * kT + t) * kH + n0 + Ncol];
      }
    }

    // GEMM: S += g[64x512] @ W_hh[512x512] restricted to our tiles
    for (int kk = 0; kk < kH; kk += 32) {
      AFrag af;  // bank-conflict-free thanks to kHp padding
      af.h[0] = *(const v8h*)&g[arow * kHp + kk + ahalf];
      af.h[1] = *(const v8h*)&g[arow * kHp + kk + 16 + ahalf];
      if (kk + 32 < kH)  // prefetch next W_hh K-slice (global_prefetch_b8)
        __builtin_prefetch(&whh16t[((cg * 4) * kH + kk + 32 + lane) * 16], 0, 1);
      #pragma unroll
      for (int j = 0; j < 4; ++j) {
        // tiled B fragment: lane holds column n = lane&15, k in A's table order
        v16h bf;
        #pragma unroll
        for (int s = 0; s < 16; ++s)
          bf[s] = whh16t[((cg * 4 + j) * kH + kk + (s & 7) + ahalf + (s >> 3) * 16) * 16 + Ncol];
        c[j] = wmma_f16(af.v, bf, c[j]);
      }
    }
    __syncthreads();  // all reads of g for this step complete

    // epilogue: sigmoid, depression update, write h/g
    #pragma unroll
    for (int j = 0; j < 4; ++j) {
      int n0 = (cg * 4 + j) * 16;
      #pragma unroll
      for (int v = 0; v < 8; ++v) {
        int b = rt * 16 + Mrow + v;
        int n = n0 + Ncol;
        float s = c[j][v];
        float h = 1.0f / (1.0f + __expf(-s));
        float y = yreg[j][v];
        y = y + 0.1f * (-(y - 1.0f) * (1.0f - h) - (y - 0.25f) * h);
        yreg[j][v] = y;
        all_hs[(b * kT + t) * kH + n] = h;   // overwrite XP slot with h_t
        g[b * kHp + n] = (_Float16)(y * h);  // gate for next step
        if (t == kT - 1) {
          h_last[b * kH + n] = h;
          y_last[b * kH + n] = y;
        }
      }
    }
    __syncthreads();  // g fully updated before next step's A-frag reads
  }
}

// ---------------------------------------------------------------------------
// Kernel 3: mean/std heads. [B*T][kH] @ [kH][16] twice, shared A fragments.
// B fragments (mwT16/swT16) are already coalesced (1KB contiguous per wave per k-step).
// ---------------------------------------------------------------------------
__global__ __launch_bounds__(256) void head_kernel(
    const float* __restrict__ all_hs,
    const _Float16* __restrict__ mwT16, const _Float16* __restrict__ swT16,
    const float* __restrict__ mean_b, const float* __restrict__ std_b,
    float* __restrict__ mean_out, float* __restrict__ std_out) {
  int gw   = (blockIdx.x * 256 + threadIdx.x) >> 5;
  int lane = threadIdx.x & 31;
  if (gw >= (kB * kT) / 16) return;
  int m0 = gw * 16;
  int r  = m0 + (lane & 15);
  int hi = lane >> 4;

  v8f cm = {}, cs = {};
  #pragma unroll 4
  for (int kk = 0; kk < kH; kk += 32) {
    AFrag af;
    const float* arow = all_hs + r * kH + kk + hi * 8;
    #pragma unroll
    for (int e = 0; e < 8; ++e) af.h[0][e] = (_Float16)arow[e];
    #pragma unroll
    for (int e = 0; e < 8; ++e) af.h[1][e] = (_Float16)arow[16 + e];
    v16h bm, bs;
    #pragma unroll
    for (int s = 0; s < 16; ++s) {
      int kb = kk + (s & 7) + hi * 8 + (s >> 3) * 16;
      bm[s] = mwT16[kb * kA + (lane & 15)];
      bs[s] = swT16[kb * kA + (lane & 15)];
    }
    cm = wmma_f16(af.v, bm, cm);
    cs = wmma_f16(af.v, bs, cs);
  }
  int Mrow = hi * 8, n = lane & 15;
  float mb = mean_b[n], sb = std_b[n];
  #pragma unroll
  for (int v = 0; v < 8; ++v) {
    int m = m0 + Mrow + v;
    mean_out[m * kA + n] = cm[v] + mb;
    float s = cs[v] + sb;
    std_out[m * kA + n] = fminf(2.0f, fmaxf(-20.0f, s));
  }
}

// ---------------------------------------------------------------------------
extern "C" void kernel_launch(void* const* d_in, const int* in_sizes, int n_in,
                              void* d_out, int out_size, void* d_ws, size_t ws_size,
                              hipStream_t stream) {
  const float* x   = (const float*)d_in[0];  // [B,T,I]
  const float* hn  = (const float*)d_in[1];  // [1,B,H]
  const float* yd  = (const float*)d_in[2];  // [1,B,H]
  const float* whh = (const float*)d_in[3];  // [H,H]
  const float* wih = (const float*)d_in[4];  // [I,H]
  const float* mw  = (const float*)d_in[5];  // [A,H]
  const float* mb  = (const float*)d_in[6];  // [A]
  const float* sw  = (const float*)d_in[7];  // [A,H]
  const float* sb  = (const float*)d_in[8];  // [A]

  float* out      = (float*)d_out;
  float* mean_out = out;                              // B*T*A = 1,024,000
  float* std_out  = out + kB * kT * kA;               // +1,024,000
  float* h_last   = out + 2 * kB * kT * kA;           // +32,768
  float* all_hs   = h_last + kB * kH;                 // B*T*H
  float* y_last   = all_hs + kB * kT * kH;            // +32,768

  _Float16* wih16t = (_Float16*)d_ws;                 // 128*512 (tiled)
  _Float16* whh16t = wih16t + kI * kH;                // 512*512 (tiled)
  _Float16* mwT16  = whh16t + kH * kH;                // 512*16
  _Float16* swT16  = mwT16 + kH * kA;                 // 512*16

  convert_kernel<<<(kH * kH) / 256, 256, 0, stream>>>(
      wih, whh, mw, sw, wih16t, whh16t, mwT16, swT16);

  // XP = x @ W_ih -> all_hs region. 4000 m-tiles * 32 n-tiles, 8 waves/block.
  xproj_kernel<<<(kB * kT / 16) * 32 / 8, 256, 0, stream>>>(x, wih16t, all_hs);

  // Serial recurrence: one workgroup, 32 waves.
  rnn_recurrence_kernel<<<1, 1024, 0, stream>>>(hn, yd, whh16t, all_hs,
                                                h_last, y_last);

  // Heads: 4000 m-tiles, 8 waves/block.
  head_kernel<<<(kB * kT / 16) / 8, 256, 0, stream>>>(
      all_hs, mwT16, swT16, mb, sb, mean_out, std_out);
}
